// VectorQuantizer_71451075936448
// MI455X (gfx1250) — matrix operand order repaired and checked by reading, compiled, not run on hardware
//
#include <hip/hip_runtime.h>

typedef __attribute__((ext_vector_type(16))) __bf16 v16bf;
typedef __attribute__((ext_vector_type(8)))  float  v8f;

#define N_ROWS         262144   // 64^4 / 64
#define DIM            64
#define K_CODES        512
#define ROWS_PER_BLOCK 128
#define THREADS        256
#define NBLOCKS        (N_ROWS / ROWS_PER_BLOCK)   // 2048

// ---------------------------------------------------------------------------
// Kernel 0: pre-swizzle embeddings into WMMA-B bf16 fragments + compute ||e||^2
//
// B fragment layout used by this code (contiguous k order, same as A; 32x16 K-major operand):
//   lane l: column = l & 15, K-half = l >> 4 ; element e -> K = 16*(l>>4) + e
// Fragment id f = tile*2 + kchunk  (tile: 16 codes, kchunk: 32 of the 64 dims)
// Stored as frag[((f*32 + lane)*16 + e)] so each lane reads 32 contiguous bytes.
// ---------------------------------------------------------------------------
__global__ void vq_prep(const float* __restrict__ E,
                        unsigned short* __restrict__ frag,
                        float* __restrict__ eb2) {
    int tid = blockIdx.x * blockDim.x + threadIdx.x;
    if (tid < 32768) {
        int e    = tid & 15;
        int lane = (tid >> 4) & 31;
        int f    = tid >> 9;              // 0..63
        int t    = f >> 1;                // code tile 0..31
        int c    = f & 1;                 // k-chunk 0..1
        int code = t * 16 + (lane & 15);
        int d    = c * 32 + (lane >> 4) * 16 + e;
        float v  = E[d * K_CODES + code];
        union { __bf16 b; unsigned short u; } cv;
        cv.b = (__bf16)v;
        frag[tid] = cv.u;
    } else if (tid < 32768 + K_CODES) {
        int k = tid - 32768;
        float s = 0.f;
        #pragma unroll 8
        for (int d = 0; d < DIM; ++d) {
            float v = E[d * K_CODES + k];
            s = fmaf(v, v, s);
        }
        eb2[k] = s;
    }
}

// ---------------------------------------------------------------------------
// Kernel 1: per-wave 16-row tile; WMMA codebook search + fused gather + loss
// ---------------------------------------------------------------------------
struct __align__(64) Smem1 {
    unsigned short bfrag[32 * 32 * 16];  // 32 KB: 32 frags x 32 lanes x 16 bf16
    float eb2[256];
    int   idx[ROWS_PER_BLOCK];
    float red[8];
};

// A layout (16x32 bf16, M-major operand): lane l holds row M = l&15,
// K-half h = l>>4; element e -> K = kbase + 16h + e
__device__ inline v16bf load_a_frag(const float* __restrict__ xr, int kbase, int half) {
    const float4* p0 = reinterpret_cast<const float4*>(xr + kbase + half * 16);
    float4 u0 = p0[0], u1 = p0[1];
    const float4* p1 = reinterpret_cast<const float4*>(xr + kbase + half * 16 + 8);
    float4 u2 = p1[0], u3 = p1[1];
    v16bf a;
    a[0]  = (__bf16)u0.x; a[1]  = (__bf16)u0.y; a[2]  = (__bf16)u0.z; a[3]  = (__bf16)u0.w;
    a[4]  = (__bf16)u1.x; a[5]  = (__bf16)u1.y; a[6]  = (__bf16)u1.z; a[7]  = (__bf16)u1.w;
    a[8]  = (__bf16)u2.x; a[9]  = (__bf16)u2.y; a[10] = (__bf16)u2.z; a[11] = (__bf16)u2.w;
    a[12] = (__bf16)u3.x; a[13] = (__bf16)u3.y; a[14] = (__bf16)u3.z; a[15] = (__bf16)u3.w;
    return a;
}

__global__ __launch_bounds__(THREADS)
void vq_main(const float* __restrict__ x,
             const unsigned short* __restrict__ fragG,
             const float* __restrict__ eb2G,
             const float* __restrict__ E,
             float* __restrict__ out,
             float* __restrict__ bsum) {
    __shared__ Smem1 s;
    const int tid  = threadIdx.x;
    const int wave = tid >> 5;
    const int lane = tid & 31;
    const int half = lane >> 4;
    const int lrow = lane & 15;

    // A fragments for this wave's 16 rows (each x byte read once per wave)
    const int rowBase = blockIdx.x * ROWS_PER_BLOCK + wave * 16;
    const float* xr = x + (size_t)(rowBase + lrow) * DIM;
    v16bf a0 = load_a_frag(xr, 0, half);
    v16bf a1 = load_a_frag(xr, 32, half);

    float best[8];
    int   bidx[8];
    #pragma unroll
    for (int r = 0; r < 8; ++r) { best[r] = 3.4e38f; bidx[r] = 0; }

    for (int p = 0; p < 2; ++p) {           // two phases of 256 codes
        __syncthreads();
        // stage 32 KB of pre-swizzled B fragments into LDS (coalesced b128)
        const int4* src = reinterpret_cast<const int4*>(fragG + p * 16384);
        int4*       dst = reinterpret_cast<int4*>(s.bfrag);
        #pragma unroll
        for (int i = 0; i < 8; ++i)
            dst[tid + i * THREADS] = src[tid + i * THREADS];
        s.eb2[tid] = eb2G[p * 256 + tid];
        __syncthreads();

        #pragma unroll 4
        for (int t = 0; t < 16; ++t) {
            v16bf b0 = *reinterpret_cast<const v16bf*>(&s.bfrag[(t * 2 + 0) * 512 + lane * 16]);
            v16bf b1 = *reinterpret_cast<const v16bf*>(&s.bfrag[(t * 2 + 1) * 512 + lane * 16]);
            v8f acc = {0.f, 0.f, 0.f, 0.f, 0.f, 0.f, 0.f, 0.f};
            acc = __builtin_amdgcn_wmma_f32_16x16x32_bf16(false, a0, false, b0,
                                                          (short)0, acc, false, false);
            acc = __builtin_amdgcn_wmma_f32_16x16x32_bf16(false, a1, false, b1,
                                                          (short)0, acc, false, false);
            float ebn = s.eb2[t * 16 + lrow];
            int  code = p * 256 + t * 16 + lrow;
            // D layout: VGPR r -> row (r + 8*half), col = lrow
            #pragma unroll
            for (int r = 0; r < 8; ++r) {
                float sc = fmaf(-2.f, acc[r], ebn);   // ||e||^2 - 2*sim
                if (sc < best[r]) { best[r] = sc; bidx[r] = code; }
            }
        }
    }

    // argmin across the 16 lanes of each half-group (first-index tie-break)
    #pragma unroll
    for (int off = 1; off < 16; off <<= 1) {
        #pragma unroll
        for (int r = 0; r < 8; ++r) {
            float ob = __shfl_xor(best[r], off, 32);
            int   oi = __shfl_xor(bidx[r], off, 32);
            if (ob < best[r] || (ob == best[r] && oi < bidx[r])) {
                best[r] = ob; bidx[r] = oi;
            }
        }
    }
    if (lrow == 0) {
        #pragma unroll
        for (int r = 0; r < 8; ++r)
            s.idx[wave * 16 + half * 8 + r] = bidx[r];   // rows r / r+8
    }
    __syncthreads();

    // fused gather (q = E[:, idx]) + straight-through output + loss partial
    float lsum = 0.f;
    const size_t outBase = (size_t)blockIdx.x * (ROWS_PER_BLOCK * DIM);
    #pragma unroll 4
    for (int j = tid; j < ROWS_PER_BLOCK * DIM; j += THREADS) {
        int m = j >> 6, d = j & 63;
        int k = s.idx[m];
        float q  = E[d * K_CODES + k];   // E is tiny -> L1/L2 resident
        float xv = x[outBase + j];       // second read, L2 resident
        out[outBase + j] = q;            // quantized_st == q numerically
        float df = q - xv;
        lsum = fmaf(df, df, lsum);
    }
    #pragma unroll
    for (int off = 16; off > 0; off >>= 1) lsum += __shfl_xor(lsum, off, 32);
    if (lane == 0) s.red[wave] = lsum;
    __syncthreads();
    if (tid == 0) {
        float t = 0.f;
        #pragma unroll
        for (int w = 0; w < 8; ++w) t += s.red[w];
        bsum[blockIdx.x] = t;
    }
}

// ---------------------------------------------------------------------------
// Kernel 2: deterministic final loss reduction (no float atomics)
// ---------------------------------------------------------------------------
__global__ void vq_loss(const float* __restrict__ bsum, float* __restrict__ loss) {
    __shared__ float r[256];
    float s = 0.f;
    for (int i = threadIdx.x; i < NBLOCKS; i += 256) s += bsum[i];
    r[threadIdx.x] = s;
    __syncthreads();
    for (int off = 128; off > 0; off >>= 1) {
        if (threadIdx.x < off) r[threadIdx.x] += r[threadIdx.x + off];
        __syncthreads();
    }
    if (threadIdx.x == 0)
        *loss = r[0] * (1.25f / 16777216.0f);   // (beta + 1) * mean
}

// ---------------------------------------------------------------------------
extern "C" void kernel_launch(void* const* d_in, const int* in_sizes, int n_in,
                              void* d_out, int out_size, void* d_ws, size_t ws_size,
                              hipStream_t stream) {
    (void)in_sizes; (void)n_in; (void)out_size; (void)ws_size;
    const float* x = (const float*)d_in[0];          // [64,64,64,64] f32
    const float* E = (const float*)d_in[1];          // [64,512] f32
    float* out  = (float*)d_out;                     // [16777216] q + [1] loss
    float* loss = out + (size_t)N_ROWS * DIM;

    unsigned short* frag = (unsigned short*)d_ws;                       // 64 KB
    float* eb2  = (float*)((char*)d_ws + 65536);                        // 2 KB
    float* bsum = (float*)((char*)d_ws + 65536 + 2048);                 // 8 KB

    vq_prep<<<130, 256, 0, stream>>>(E, frag, eb2);
    vq_main<<<NBLOCKS, THREADS, 0, stream>>>(x, frag, eb2, E, out, bsum);
    vq_loss<<<1, 256, 0, stream>>>(bsum, loss);
}
